// PostProcessWrapper_25606595018881
// MI455X (gfx1250) — 8-bit matrix operand order repaired and checked by reading, compiled, not run on hardware
//
#include <hip/hip_runtime.h>
#include <hip/hip_bf16.h>

#define BATCH    32
#define NANCH    8400
#define NCLS     80
#define NFLAT    (NANCH * NCLS)
#define PRE_K    1024
#define MAXDET   300
#define SCORE_TH 0.001f
#define IOU_THR  0.7f

#define NTHR     1024
#define CANDCAP  2048
#define NBINS    16384
#define CHUNK    (NBINS / NTHR)

typedef int v8i __attribute__((ext_vector_type(8)));
typedef unsigned long long u64;

__global__ __launch_bounds__(NTHR) void topk_select(
    const float* __restrict__ boxes,
    const float* __restrict__ scores,
    float* __restrict__ cboxes,
    float* __restrict__ cscores,
    int*   __restrict__ clabels,
    int*   __restrict__ cvalid)
{
    __shared__ unsigned int bins[NBINS];
    __shared__ unsigned int part[NTHR];
    __shared__ u64 cand[CANDCAP];
    __shared__ unsigned int s_cnt, s_cb, s_above, s_cut;

    const int b = blockIdx.x;
    const int t = threadIdx.x;
    const float* sc = scores + (size_t)b * NFLAT;

    for (int i = t; i < NBINS; i += NTHR) bins[i] = 0u;
    for (int i = t; i < CANDCAP; i += NTHR) cand[i] = 0ull;
    if (t == 0) s_cnt = 0u;
    __syncthreads();

    for (int i = t; i < NFLAT; i += NTHR) {
        __builtin_prefetch(sc + i + 4 * NTHR, 0, 1);
        float v = sc[i];
        if (v > SCORE_TH) {
            unsigned bits = __float_as_uint(v);
            atomicAdd(&bins[bits >> 16], 1u);
        }
    }
    __syncthreads();

    {
        unsigned p = 0;
        for (int k = 0; k < CHUNK; ++k) p += bins[t * CHUNK + k];
        part[t] = p;
    }
    __syncthreads();
    if (t == 0) {
        unsigned acc = 0, above = 0; int cb = -1;
        for (int c = NTHR - 1; c >= 0; --c) {
            unsigned p = part[c];
            if (acc + p >= PRE_K) {
                for (int i = c * CHUNK + CHUNK - 1; i >= c * CHUNK; --i) {
                    unsigned cnt = bins[i];
                    if (acc + cnt >= PRE_K) { cb = i; above = acc; break; }
                    acc += cnt;
                }
                break;
            }
            acc += p;
        }
        if (cb < 0) { s_cb = 0xFFFFFFFFu; s_above = acc; }
        else        { s_cb = (unsigned)cb; s_above = above; }
    }
    __syncthreads();
    const unsigned cb = s_cb;

    for (int i = t; i < NBINS; i += NTHR) bins[i] = 0u;
    __syncthreads();
    if (cb != 0xFFFFFFFFu) {
        for (int i = t; i < NFLAT; i += NTHR) {
            __builtin_prefetch(sc + i + 4 * NTHR, 0, 1);
            float v = sc[i];
            if (v > SCORE_TH) {
                unsigned bits = __float_as_uint(v);
                if ((bits >> 16) == cb) atomicAdd(&bins[(bits >> 2) & (NBINS - 1)], 1u);
            }
        }
    }
    __syncthreads();
    {
        unsigned p = 0;
        for (int k = 0; k < CHUNK; ++k) p += bins[t * CHUNK + k];
        part[t] = p;
    }
    __syncthreads();
    if (t == 0) {
        if (cb == 0xFFFFFFFFu) s_cut = 0u;
        else {
            unsigned acc = s_above, fb = 0;
            for (int c = NTHR - 1; c >= 0; --c) {
                unsigned p = part[c];
                if (acc + p >= PRE_K) {
                    for (int i = c * CHUNK + CHUNK - 1; i >= c * CHUNK; --i) {
                        acc += bins[i];
                        if (acc >= PRE_K) { fb = (unsigned)i; break; }
                    }
                    break;
                }
                acc += p;
            }
            s_cut = (cb << 14) | fb;
        }
    }
    __syncthreads();
    const unsigned cut = s_cut;

    for (int i = t; i < NFLAT; i += NTHR) {
        float v = sc[i];
        if (v > SCORE_TH) {
            unsigned bits = __float_as_uint(v);
            if ((bits >> 2) >= cut) {
                unsigned p = atomicAdd(&s_cnt, 1u);
                if (p < CANDCAP)
                    cand[p] = ((u64)bits << 32) | (u64)(0xFFFFFFFFu - (unsigned)i);
            }
        }
    }
    __syncthreads();

    for (unsigned kk = 2; kk <= CANDCAP; kk <<= 1) {
        for (unsigned j = kk >> 1; j > 0; j >>= 1) {
            __syncthreads();
            for (unsigned i = t; i < CANDCAP; i += NTHR) {
                unsigned ixj = i ^ j;
                if (ixj > i) {
                    u64 a = cand[i], c = cand[ixj];
                    bool desc = ((i & kk) == 0);
                    if ((a < c) == desc) { cand[i] = c; cand[ixj] = a; }
                }
            }
        }
    }
    __syncthreads();

    if (t < PRE_K) {
        u64 key = cand[t];
        float sv; int lbl, vld; float b0, b1, b2, b3;
        if (key != 0ull) {
            unsigned bits = (unsigned)(key >> 32);
            unsigned e = 0xFFFFFFFFu - (unsigned)key;
            sv  = __uint_as_float(bits);
            lbl = (int)(e % NCLS);
            vld = 1;
            const float* bp = boxes + ((size_t)b * NANCH + e / NCLS) * 4;
            b0 = bp[0]; b1 = bp[1]; b2 = bp[2]; b3 = bp[3];
        } else {
            sv = -1.0f; lbl = 0; vld = 0; b0 = b1 = b2 = b3 = 0.0f;
        }
        size_t o = (size_t)b * PRE_K + t;
        cscores[o] = sv; clabels[o] = lbl; cvalid[o] = vld;
        float* cb4 = cboxes + o * 4;
        cb4[0] = b0; cb4[1] = b1; cb4[2] = b2; cb4[3] = b3;
    }
}

__global__ __launch_bounds__(256) void mask_kernel(
    const float* __restrict__ cboxes,
    const int*   __restrict__ clabels,
    const int*   __restrict__ cvalid,
    u64*         __restrict__ mask)
{
    const int b    = blockIdx.y;
    const int wave = threadIdx.x >> 5;
    const int lane = threadIdx.x & 31;
    const int it   = blockIdx.x * 8 + wave;
    const int i0   = it * 16;
    const int half = lane >> 4;
    const int l16  = lane & 15;

    const float* CB = cboxes + (size_t)b * PRE_K * 4;
    const int*   CL = clabels + (size_t)b * PRE_K;
    const int*   CV = cvalid  + (size_t)b * PRE_K;

    const int ca = CL[i0 + l16];
    union { v8i v; int e[8]; } A0, A1;
#pragma unroll
    for (int v = 0; v < 8; ++v) {
        int ks = ((v >= 4) ? 32 : 0) + (((v & 3) >= 2) ? 16 : 0) + ((v & 1) * 4) + half * 8;
        int c0 = ca - ks;
        A0.e[v] = (c0 >= 0 && c0 < 4) ? (1 << (c0 * 8)) : 0;
        int c1 = ca - 64 - ks;
        A1.e[v] = (c1 >= 0 && c1 < 4) ? (1 << (c1 * 8)) : 0;
    }

    float ax1[8], ay1[8], ax2[8], ay2[8], aar[8];
#pragma unroll
    for (int r = 0; r < 8; ++r) {
        const float* bp = CB + (size_t)(i0 + r + 8 * half) * 4;
        ax1[r] = bp[0]; ay1[r] = bp[1]; ax2[r] = bp[2]; ay2[r] = bp[3];
        aar[r] = fmaxf(ax2[r] - ax1[r], 0.0f) * fmaxf(ay2[r] - ay1[r], 0.0f);
    }

    const int w0 = it >> 2;
    if (lane < 16) {
        for (int w = 0; w < w0; ++w)
            mask[((size_t)b * PRE_K + i0 + lane) * 16 + w] = 0ull;
    }

    u64 accum = 0ull;
    for (int jt = w0 * 4; jt < 64; ++jt) {
        const int j = jt * 16 + l16;
        const int cj = CL[j];
        union { v8i v; int e[8]; } B0, B1;
#pragma unroll
        for (int v = 0; v < 8; ++v) {
            int ks = ((v >= 4) ? 32 : 0) + (((v & 3) >= 2) ? 16 : 0) + ((v & 1) * 4) + half * 8;
            int c0 = cj - ks;
            B0.e[v] = (c0 >= 0 && c0 < 4) ? (1 << (c0 * 8)) : 0;
            int c1 = cj - 64 - ks;
            B1.e[v] = (c1 >= 0 && c1 < 4) ? (1 << (c1 * 8)) : 0;
        }

        v8i acc = {};
        acc = __builtin_amdgcn_wmma_i32_16x16x64_iu8(false, A0.v, false, B0.v, acc, false, false);
        acc = __builtin_amdgcn_wmma_i32_16x16x64_iu8(false, A1.v, false, B1.v, acc, false, false);
        union { v8i v; int e[8]; } D; D.v = acc;

        const float* jp = CB + (size_t)j * 4;
        const float jx1 = jp[0], jy1 = jp[1], jx2 = jp[2], jy2 = jp[3];
        const float jar = fmaxf(jx2 - jx1, 0.0f) * fmaxf(jy2 - jy1, 0.0f);
        const int jvalid = CV[j];

        unsigned row16 = 0;
#pragma unroll
        for (int r = 0; r < 8; ++r) {
            const int i = i0 + r + 8 * half;
            float ix1 = fmaxf(ax1[r], jx1), iy1 = fmaxf(ay1[r], jy1);
            float ix2 = fminf(ax2[r], jx2), iy2 = fminf(ay2[r], jy2);
            float inter = fmaxf(ix2 - ix1, 0.0f) * fmaxf(iy2 - iy1, 0.0f);
            float uni = aar[r] + jar - inter;
            bool over = inter > IOU_THR * fmaxf(uni, 1e-9f);
            bool bit = (j > i) && (D.e[r] != 0) && (jvalid != 0) && over;
            unsigned blt = (unsigned)__ballot(bit);
            if (lane == r)            row16 = blt & 0xFFFFu;
            else if (lane == (r + 8)) row16 = (blt >> 16) & 0xFFFFu;
        }
        if (lane < 16) {
            accum |= (u64)row16 << (16 * (jt & 3));
            if ((jt & 3) == 3) {
                mask[((size_t)b * PRE_K + i0 + lane) * 16 + (jt >> 2)] = accum;
                accum = 0ull;
            }
        }
    }
}

__global__ __launch_bounds__(NTHR) void nms_kernel(
    const float* __restrict__ cboxes,
    const float* __restrict__ cscores,
    const int*   __restrict__ clabels,
    const int*   __restrict__ cvalid,
    const u64*   __restrict__ mask,
    float*       __restrict__ out)
{
    __shared__ u64 s_keep[16];
    const int b = blockIdx.x;
    const int t = threadIdx.x;
    const int* CV = cvalid + (size_t)b * PRE_K;
    const u64* M  = mask + (size_t)b * PRE_K * 16;

    if (t < 32) {
        const int lane = t;
        u64 sup = 0ull, keep = 0ull;
        for (int i = 0; i < PRE_K; ++i) {
            const int owner = i >> 6;
            u64 supw = __shfl(sup, owner, 32);
            bool act = (((supw >> (i & 63)) & 1ull) == 0ull) && (CV[i] != 0);
            if (act) {
                if (lane < 16) sup |= M[(size_t)i * 16 + lane];
                if (lane == owner) keep |= (1ull << (i & 63));
            }
        }
        if (lane < 16) s_keep[lane] = keep;
    }
    __syncthreads();

    float* ob = out;
    float* os = out + (size_t)BATCH * MAXDET * 4;
    float* ol = os + (size_t)BATCH * MAXDET;
    float* on = ol + (size_t)BATCH * MAXDET;

    if (t < MAXDET) {
        size_t o = (size_t)b * MAXDET + t;
        ob[o * 4 + 0] = 0.0f; ob[o * 4 + 1] = 0.0f;
        ob[o * 4 + 2] = 0.0f; ob[o * 4 + 3] = 0.0f;
        os[o] = 0.0f; ol[o] = 0.0f;
    }
    __syncthreads();

    bool kept = (s_keep[t >> 6] >> (t & 63)) & 1ull;
    int rank = 0;
    for (int w = 0; w < (t >> 6); ++w) rank += __popcll(s_keep[w]);
    rank += __popcll(s_keep[t >> 6] & ((1ull << (t & 63)) - 1ull));

    if (kept && rank < MAXDET) {
        size_t src = (size_t)b * PRE_K + t;
        size_t dst = (size_t)b * MAXDET + rank;
        const float* bp = cboxes + src * 4;
        ob[dst * 4 + 0] = bp[0]; ob[dst * 4 + 1] = bp[1];
        ob[dst * 4 + 2] = bp[2]; ob[dst * 4 + 3] = bp[3];
        os[dst] = cscores[src];
        ol[dst] = (float)clabels[src];
    }
    if (t == 0) {
        int tot = 0;
        for (int w = 0; w < 16; ++w) tot += __popcll(s_keep[w]);
        on[b] = (float)(tot < MAXDET ? tot : MAXDET);
    }
}

extern "C" void kernel_launch(void* const* d_in, const int* in_sizes, int n_in,
                              void* d_out, int out_size, void* d_ws, size_t ws_size,
                              hipStream_t stream) {
    const float* boxes  = (const float*)d_in[0];
    const float* scores = (const float*)d_in[1];
    (void)in_sizes; (void)n_in; (void)out_size; (void)ws_size;

    float* cboxes  = (float*)d_ws;
    float* cscores = cboxes + (size_t)BATCH * PRE_K * 4;
    int*   clabels = (int*)(cscores + (size_t)BATCH * PRE_K);
    int*   cvalid  = clabels + (size_t)BATCH * PRE_K;
    u64*   mask    = (u64*)(cvalid + (size_t)BATCH * PRE_K);

    topk_select<<<BATCH, NTHR, 0, stream>>>(boxes, scores, cboxes, cscores, clabels, cvalid);
    mask_kernel<<<dim3(64 / 8, BATCH), 256, 0, stream>>>(cboxes, clabels, cvalid, mask);
    nms_kernel<<<BATCH, NTHR, 0, stream>>>(cboxes, cscores, clabels, cvalid, mask, (float*)d_out);
}
